// ConditionalRandomField_77747497992502
// MI455X (gfx1250) — matrix operand order repaired and checked by reading, not compiled, not run on hardware
//
#include <hip/hip_runtime.h>
#include <hip/hip_bf16.h>
#include <math.h>

// ---- problem constants (from reference) ----
#define NT    256     // NUM_TAGS
#define SEQL  1024    // SEQ
#define NB    64      // BATCH

typedef __attribute__((ext_vector_type(16))) _Float16 v16h;
typedef __attribute__((ext_vector_type(8)))  _Float16 v8h;
typedef __attribute__((ext_vector_type(8)))  float    v8f;

// ============================================================================
// Forward-algorithm normalizer.
//   grid = 4 blocks (16 batches each), block = 512 threads = 16 wave32s.
//   Wave w owns N-columns [16w,16w+16). E=exp(T) lives in registers as f16
//   B-fragments. Per step: rescale+exp score rows into LDS (f16 A tile),
//   8x v_wmma_f32_16x16x32_f16 (K=256), then score[j] = em + log(q) + m.
// ============================================================================
__global__ __launch_bounds__(512)
void crf_forward_kernel(const float* __restrict__ em,
                        const float* __restrict__ T,
                        const float* __restrict__ startT,
                        const float* __restrict__ endT,
                        const unsigned char* __restrict__ mask,
                        float* __restrict__ nrmOut)
{
    __shared__ float    scoreLds[16 * NT];                       // 16 KB
    __shared__ _Float16 Plds[16 * NT] __attribute__((aligned(16))); // 8 KB
    __shared__ float    mLds[16];

    const int tid    = threadIdx.x;
    const int w      = tid >> 5;          // wave id 0..15  (owns N-tile w and batch-row w)
    const int lane   = tid & 31;
    const int bg0    = blockIdx.x * 16;   // first batch of this block
    const int Nl     = lane & 15;
    const int hiHalf = lane >> 4;         // 0 for lanes 0-15, 1 for lanes 16-31
    const int jcol   = w * 16 + Nl;       // this lane's output column (tag index)

    // ---- Build resident B fragments: E[k][j] = exp(T[k][j]) as f16 --------
    // B k order used here (contiguous, same as A): lanes 0-15 -> N=lane, K = kbase+0..15 ; lanes 16-31: K += 16
    v16h bfrag[8];
#pragma unroll
    for (int kb = 0; kb < 8; ++kb) {
        const int kb0 = kb * 32 + hiHalf * 16;
#pragma unroll
        for (int h = 0; h < 16; ++h)
            bfrag[kb][h] = (_Float16)__expf(T[(kb0 + h) * NT + jcol]);
    }

    // ---- score0[b][j] = start[j] + em[b][0][j] ----------------------------
    for (int j = lane; j < NT; j += 32)
        scoreLds[w * NT + j] = startT[j] + em[(size_t)(bg0 + w) * SEQL * NT + j];

    for (int t = 1; t < SEQL; ++t) {
        __syncthreads();   // phase-4 writes of previous step complete

        // issue global loads for this step early (overlap with reductions/WMMA)
        float emv[8];
        int   mskv[8];
#pragma unroll
        for (int r = 0; r < 8; ++r) {
            const int M = r + hiHalf * 8;
            const int b = bg0 + M;
            emv[r]  = em[(size_t)b * SEQL * NT + (size_t)t * NT + jcol];
            mskv[r] = (int)mask[b * SEQL + t];
        }

        // phase 1: row max (wave w reduces its own batch row w)
        float mm = -INFINITY;
        for (int j = lane; j < NT; j += 32)
            mm = fmaxf(mm, scoreLds[w * NT + j]);
#pragma unroll
        for (int off = 16; off > 0; off >>= 1)
            mm = fmaxf(mm, __shfl_xor(mm, off, 32));
        if (lane == 0) mLds[w] = mm;

        // phase 2: P[w][k] = exp(score - m) as f16 (A tile)
        for (int j = lane; j < NT; j += 32)
            Plds[w * NT + j] = (_Float16)__expf(scoreLds[w * NT + j] - mm);

        __syncthreads();   // P + mLds visible to all waves

        // phase 3: Q-tile = P(16x256) @ E(256x16w-tile), f32 accumulate
        v8f c = {0.f, 0.f, 0.f, 0.f, 0.f, 0.f, 0.f, 0.f};
        const int Ma = Nl;                          // A-matrix row = lane&15
#pragma unroll
        for (int kb = 0; kb < 8; ++kb) {
            // A k order used here (contiguous, same as B): lanes 0-15 -> K = kb*32 + 0..15
            //                      lanes 16-31 -> K = kb*32 + 16..31
            const int ka = kb * 32 + hiHalf * 16;
            const _Float16* pa = &Plds[Ma * NT + ka];
            v16h a;
            *(v8h*)&a       = *(const v8h*)pa;         // ds_load_b128
            *((v8h*)&a + 1) = *(const v8h*)(pa + 8);   // ds_load_b128
            c = __builtin_amdgcn_wmma_f32_16x16x32_f16(
                    false, a, false, bfrag[kb], (short)0, c, false, false);
        }

        // phase 4: score[M][jcol] = em + log(q) + m  (where mask)
        // C/D layout: reg r, lanes 0-15 -> M=r, N=lane ; lanes 16-31 -> M=r+8
#pragma unroll
        for (int r = 0; r < 8; ++r) {
            const int M  = r + hiHalf * 8;
            const float nv = emv[r] + __logf(c[r]) + mLds[M];
            if (mskv[r]) scoreLds[M * NT + jcol] = nv;
        }
    }

    __syncthreads();
    // ---- normalizer[b] = logsumexp_j(score[b][j] + end[j]) ----------------
    float m2 = -INFINITY;
    for (int j = lane; j < NT; j += 32)
        m2 = fmaxf(m2, scoreLds[w * NT + j] + endT[j]);
#pragma unroll
    for (int off = 16; off > 0; off >>= 1)
        m2 = fmaxf(m2, __shfl_xor(m2, off, 32));
    float s = 0.f;
    for (int j = lane; j < NT; j += 32)
        s += __expf(scoreLds[w * NT + j] + endT[j] - m2);
#pragma unroll
    for (int off = 16; off > 0; off >>= 1)
        s += __shfl_xor(s, off, 32);
    if (lane == 0) nrmOut[bg0 + w] = m2 + __logf(s);
}

// ============================================================================
// Gold-path score: gathers along tags. One block per batch.
// ============================================================================
__global__ __launch_bounds__(256)
void crf_score_kernel(const float* __restrict__ em,
                      const float* __restrict__ T,
                      const float* __restrict__ startT,
                      const float* __restrict__ endT,
                      const int* __restrict__ tags,
                      const unsigned char* __restrict__ mask,
                      float* __restrict__ scOut)
{
    const int b   = blockIdx.x;
    const int tid = threadIdx.x;
    float part = 0.f;
    int   cnt  = 0;
    for (int t = tid; t < SEQL; t += 256) {
        const int mk = mask[b * SEQL + t] ? 1 : 0;
        cnt += mk;
        if (t >= 1 && mk) {
            const int ct = tags[b * SEQL + t];
            const int pt = tags[b * SEQL + t - 1];
            part += T[ct * NT + pt] +
                    em[(size_t)b * SEQL * NT + (size_t)t * NT + ct];
        }
    }
    __shared__ float sf[256];
    __shared__ int   si[256];
    sf[tid] = part; si[tid] = cnt;
    __syncthreads();
    for (int off = 128; off > 0; off >>= 1) {
        if (tid < off) { sf[tid] += sf[tid + off]; si[tid] += si[tid + off]; }
        __syncthreads();
    }
    if (tid == 0) {
        const int t0   = tags[b * SEQL];
        const int send = si[0] - 1;
        const int lt   = tags[b * SEQL + send];
        scOut[b] = startT[t0] + em[(size_t)b * SEQL * NT + t0] + sf[0] + endT[lt];
    }
}

// ============================================================================
// mean(normalizer - score)
// ============================================================================
__global__ __launch_bounds__(64)
void crf_final_kernel(const float* __restrict__ nrm,
                      const float* __restrict__ sc,
                      float* __restrict__ out)
{
    __shared__ float s[64];
    const int tid = threadIdx.x;
    s[tid] = nrm[tid] - sc[tid];
    __syncthreads();
    for (int off = 32; off > 0; off >>= 1) {
        if (tid < off) s[tid] += s[tid + off];
        __syncthreads();
    }
    if (tid == 0) out[0] = s[0] / (float)NB;
}

// ============================================================================
extern "C" void kernel_launch(void* const* d_in, const int* in_sizes, int n_in,
                              void* d_out, int out_size, void* d_ws, size_t ws_size,
                              hipStream_t stream)
{
    const float*         em     = (const float*)d_in[0];          // (64,1024,256)
    const float*         T      = (const float*)d_in[1];          // (256,256)
    const float*         startT = (const float*)d_in[2];          // (256,)
    const float*         endT   = (const float*)d_in[3];          // (256,)
    const int*           tags   = (const int*)d_in[4];            // (64,1024) int32
    const unsigned char* mask   = (const unsigned char*)d_in[5];  // (64,1024) bool

    float* ws  = (float*)d_ws;
    float* nrm = ws;        // 64 floats
    float* sc  = ws + 64;   // 64 floats

    crf_forward_kernel<<<dim3(4),  dim3(512), 0, stream>>>(em, T, startT, endT, mask, nrm);
    crf_score_kernel  <<<dim3(64), dim3(256), 0, stream>>>(em, T, startT, endT, tags, mask, sc);
    crf_final_kernel  <<<dim3(1),  dim3(64),  0, stream>>>(nrm, sc, (float*)d_out);
}
